// AttentionGNNLayer_31834297598228
// MI455X (gfx1250) — matrix operand order repaired and checked by reading, not compiled, not run on hardware
//
#include <hip/hip_runtime.h>
#include <hip/hip_bf16.h>

// ---------------------------------------------------------------------------
// CDNA5 (gfx1250) AttentionGNNLayer
//   - bf16 WMMA (v_wmma_f32_16x16x32_bf16), f32 accumulate
//   - A fragments: 2x global_load_b128 (contiguous K-run per lane, same order as B swizzle)
//   - B fragments: pre-swizzled weights -> 2x ds_load_b128 per fragment
//   - L2-resident segment softmax + scatter via f32 atomics
// ---------------------------------------------------------------------------

#define D 128

typedef __bf16 v16bf __attribute__((ext_vector_type(16)));
typedef __bf16 v8bf  __attribute__((ext_vector_type(8)));
typedef float  v8f   __attribute__((ext_vector_type(8)));

union frag_u {
  v16bf v;
  struct { v8bf lo; v8bf hi; } h;
};

static __device__ __forceinline__ unsigned short f2bf(float f) {
  unsigned int u = __float_as_uint(f);
  u += 0x7FFFu + ((u >> 16) & 1u);      // round-to-nearest-even
  return (unsigned short)(u >> 16);
}

static __device__ __forceinline__ v8f vzero8() {
  v8f z;
#pragma unroll
  for (int i = 0; i < 8; ++i) z[i] = 0.f;
  return z;
}

static __device__ __forceinline__ v8f wmma_bf16(v16bf a, v16bf b, v8f c) {
  // (neg_a, A, neg_b, B, c_mod, C, reuse_a, reuse_b)
  return __builtin_amdgcn_wmma_f32_16x16x32_bf16(false, a, false, b, (short)0, c,
                                                 false, false);
}

// A fragment: 16x32 bf16, M = lane&15.  Per-lane elements are one
// contiguous K-run [kb, kb+16) matching the B swizzle
// (kb = 0 for lanes 0-15, 16 for lanes 16-31)  ->  2x 16-byte loads.
static __device__ __forceinline__ v16bf load_a_frag(const unsigned short* __restrict__ row,
                                                    int k0, int lane) {
  const int kb = (lane & 16) ? 16 : 0;
  frag_u u;
  u.h.lo = *(const v8bf*)(row + k0 + kb);
  u.h.hi = *(const v8bf*)(row + k0 + kb + 8);
  return u.v;
}

// B fragment from pre-swizzled weights: one contiguous 32-byte block per lane.
static __device__ __forceinline__ v16bf load_b_frag(const unsigned short* __restrict__ sw,
                                                    int kt, int nt, int lane) {
  const unsigned short* p = sw + (((kt * 8 + nt) * 32 + lane) << 4);
  frag_u u;
  u.h.lo = *(const v8bf*)(p);
  u.h.hi = *(const v8bf*)(p + 8);
  return u.v;
}

// ---------------------------------------------------------------------------
// Prep kernels
// ---------------------------------------------------------------------------
__global__ void cvt_bf16_kernel(const float* __restrict__ in,
                                unsigned short* __restrict__ out, int n) {
  for (int i = blockIdx.x * blockDim.x + threadIdx.x; i < n;
       i += gridDim.x * blockDim.x)
    out[i] = f2bf(in[i]);
}

// Swizzle a [K][128] f32 weight matrix into bf16 WMMA B-fragment order:
// out[((kt*8+nt)*32+lane)*16 + i] = W[kt*32 + kb16 + i][nt*16 + (lane&15)]
// with kb16 = 0 (lanes 0-15) or 16 (lanes 16-31).
__global__ void swizzle_w_kernel(const float* __restrict__ W,
                                 unsigned short* __restrict__ out, int K) {
  const int total = K * D;
  for (int idx = blockIdx.x * blockDim.x + threadIdx.x; idx < total;
       idx += gridDim.x * blockDim.x) {
    const int i    = idx & 15;
    const int lane = (idx >> 4) & 31;
    const int nt   = (idx >> 9) & 7;
    const int kt   = idx >> 12;
    const int kb16 = (lane & 16) ? 16 : 0;
    const int k = kt * 32 + kb16 + i;
    const int n = nt * 16 + (lane & 15);
    out[idx] = f2bf(W[k * D + n]);
  }
}

__global__ void zero_kernel(float* __restrict__ p, int n) {
  for (int i = blockIdx.x * blockDim.x + threadIdx.x; i < n;
       i += gridDim.x * blockDim.x)
    p[i] = 0.f;
}

// pass 2 of segment softmax: exp(e - max) and segment sums
__global__ void expsum_kernel(const float* __restrict__ segmax,
                              float* __restrict__ evals,
                              float* __restrict__ segsum,
                              const int* __restrict__ src, int E) {
  const int i = blockIdx.x * blockDim.x + threadIdx.x;
  if (i < E) {
    const int s = src[i];
    const float v = __expf(evals[i] - segmax[s]);
    evals[i] = v;
    atomicAdd(segsum + s, v);
  }
}

// ---------------------------------------------------------------------------
// ER attention logits: e_ij = tanh([r|h] @ W_attn + b) . w0 + w0_b
// 16 edges / wave; swizzled W_attn (256x128) streamed via LDS in two halves.
// ---------------------------------------------------------------------------
__global__ void __launch_bounds__(256)
er_logits_kernel(const unsigned short* __restrict__ node_bf,
                 const int* __restrict__ er_src, const int* __restrict__ er_dst,
                 int E,
                 const unsigned short* __restrict__ Wa_sw,   // swizzled 256x128
                 const float* __restrict__ Wa_b,
                 const float* __restrict__ w0_w, const float* __restrict__ w0_b,
                 float* __restrict__ evals, float* __restrict__ segmax) {
  __shared__ __attribute__((aligned(32))) unsigned short sW[D * D];  // 32 KB
  const int lane = threadIdx.x & 31;
  const int wv   = threadIdx.x >> 5;
  const int tb   = (blockIdx.x * 8 + wv) * 16;

  v8f acc[8];
#pragma unroll
  for (int nt = 0; nt < 8; ++nt) acc[nt] = vzero8();

  int emc = tb + (lane & 15);
  if (emc >= E) emc = E - 1;

  for (int half = 0; half < 2; ++half) {
    // cooperative LDS fill of this (swizzled) K-half of W_attn
    const uint4* gsrc = (const uint4*)(Wa_sw + half * D * D);
    uint4* sdst = (uint4*)sW;
    for (int i = threadIdx.x; i < (D * D) / 8; i += blockDim.x) sdst[i] = gsrc[i];
    __syncthreads();

    // K 0..127 is r_emb (er_dst row), K 128..255 is h_emb (er_src row)
    const int grow = (half == 0) ? er_dst[emc] : er_src[emc];
    const unsigned short* rowp = node_bf + (size_t)grow * D;
    __builtin_prefetch(rowp, 0, 0);          // global_prefetch_b8

    if (tb < E) {                            // wave-uniform: EXEC all-1 for WMMA
#pragma unroll
      for (int kt = 0; kt < 4; ++kt) {
        v16bf a = load_a_frag(rowp, kt * 32, lane);
#pragma unroll
        for (int nt = 0; nt < 8; ++nt) {
          v16bf b = load_b_frag(sW, kt, nt, lane);
          acc[nt] = wmma_bf16(a, b, acc[nt]);
        }
      }
    }
    __syncthreads();
  }
  if (tb >= E) return;

  // epilogue: lane holds C[m][n], n = nt*16 + (lane&15), m = v + 8*(lane>>4)
  float part[8];
#pragma unroll
  for (int v = 0; v < 8; ++v) part[v] = 0.f;
#pragma unroll
  for (int nt = 0; nt < 8; ++nt) {
    const int n  = nt * 16 + (lane & 15);
    const float bn = Wa_b[n];
    const float wn = w0_w[n];
#pragma unroll
    for (int v = 0; v < 8; ++v)
      part[v] += tanhf(acc[nt][v] + bn) * wn;
  }
  // reduce over n (16 lanes per half): xor masks < 16 keep halves separate
#pragma unroll
  for (int v = 0; v < 8; ++v) {
    part[v] += __shfl_xor(part[v], 1, 32);
    part[v] += __shfl_xor(part[v], 2, 32);
    part[v] += __shfl_xor(part[v], 4, 32);
    part[v] += __shfl_xor(part[v], 8, 32);
  }
  if ((lane & 15) == 0) {
    const int mh = (lane >> 4) * 8;
    const float b0 = w0_b[0];
#pragma unroll
    for (int v = 0; v < 8; ++v) {
      const int e = tb + mh + v;
      if (e < E) {
        const float eij = part[v] + b0;
        evals[e] = eij;
        // segmax init 0 (reference takes max with 0); compared values are
        // >= 0, so int-bit atomicMax is order-correct for IEEE floats.
        if (eij > 0.f)
          atomicMax((int*)(segmax + er_src[e]), __float_as_int(eij));
      }
    }
  }
}

// ---------------------------------------------------------------------------
// Scatter messages: agg[sidx[e]] += weight(e) * (node[gidx[e]] @ W_neigh + b)
// mode 0: weight = exp_val / (segsum + 1e-9)   (ER edges)
// mode 1: weight = ee_weight                    (EE edges)
// ---------------------------------------------------------------------------
__global__ void __launch_bounds__(256)
scatter_msg_kernel(const unsigned short* __restrict__ node_bf,
                   const int* __restrict__ gidx, const int* __restrict__ sidx,
                   int E, int mode,
                   const float* __restrict__ evals,
                   const float* __restrict__ segsum,
                   const float* __restrict__ ew,
                   const unsigned short* __restrict__ Wn_sw,  // swizzled 128x128
                   const float* __restrict__ Wn_b,
                   float* __restrict__ agg) {
  __shared__ __attribute__((aligned(32))) unsigned short sW[D * D];  // 32 KB
  __shared__ float sWt[8][16];
  __shared__ int   sDst[8][16];
  {
    const uint4* gsrc = (const uint4*)Wn_sw;
    uint4* sdst = (uint4*)sW;
    for (int i = threadIdx.x; i < (D * D) / 8; i += blockDim.x) sdst[i] = gsrc[i];
  }
  __syncthreads();

  const int lane = threadIdx.x & 31;
  const int wv   = threadIdx.x >> 5;
  const int tb   = (blockIdx.x * 8 + wv) * 16;
  if (tb >= E) return;                       // wave-uniform; no barriers after

  int emc = tb + (lane & 15);
  if (emc >= E) emc = E - 1;
  const int grow = gidx[emc];
  if (lane < 16) {                           // stage per-edge weight + dst node
    const int dnode = sidx[emc];
    float wt = (mode == 0) ? evals[emc] / (segsum[dnode] + 1e-9f) : ew[emc];
    sWt[wv][lane]  = (tb + lane < E) ? wt : 0.f;
    sDst[wv][lane] = dnode;
  }
  const unsigned short* rowp = node_bf + (size_t)grow * D;
  __builtin_prefetch(rowp, 0, 0);

  v8f acc[8];
#pragma unroll
  for (int nt = 0; nt < 8; ++nt) acc[nt] = vzero8();
#pragma unroll
  for (int kt = 0; kt < 4; ++kt) {
    v16bf a = load_a_frag(rowp, kt * 32, lane);
#pragma unroll
    for (int nt = 0; nt < 8; ++nt) {
      v16bf b = load_b_frag(sW, kt, nt, lane);
      acc[nt] = wmma_bf16(a, b, acc[nt]);
    }
  }

  const int mh = (lane >> 4) * 8;
#pragma unroll
  for (int nt = 0; nt < 8; ++nt) {
    const int n  = nt * 16 + (lane & 15);
    const float bn = Wn_b[n];
#pragma unroll
    for (int v = 0; v < 8; ++v) {
      const int mm = mh + v;
      if (tb + mm < E) {
        const float val = (acc[nt][v] + bn) * sWt[wv][mm];
        atomicAdd(agg + (size_t)sDst[wv][mm] * D + n, val);  // L2 atomic add
      }
    }
  }
}

// ---------------------------------------------------------------------------
// Final: out = tanh(node_emb @ W_self + b_self + agg)
// ---------------------------------------------------------------------------
__global__ void __launch_bounds__(256)
final_kernel(const unsigned short* __restrict__ node_bf,
             const unsigned short* __restrict__ Ws_sw,       // swizzled 128x128
             const float* __restrict__ Ws_b,
             const float* __restrict__ agg, float* __restrict__ out, int Nn) {
  __shared__ __attribute__((aligned(32))) unsigned short sW[D * D];
  {
    const uint4* gsrc = (const uint4*)Ws_sw;
    uint4* sdst = (uint4*)sW;
    for (int i = threadIdx.x; i < (D * D) / 8; i += blockDim.x) sdst[i] = gsrc[i];
  }
  __syncthreads();

  const int lane = threadIdx.x & 31;
  const int wv   = threadIdx.x >> 5;
  const int tb   = (blockIdx.x * 8 + wv) * 16;
  if (tb >= Nn) return;

  int rowc = tb + (lane & 15);
  if (rowc >= Nn) rowc = Nn - 1;
  const unsigned short* rowp = node_bf + (size_t)rowc * D;
  __builtin_prefetch(rowp, 0, 0);

  v8f acc[8];
#pragma unroll
  for (int nt = 0; nt < 8; ++nt) acc[nt] = vzero8();
#pragma unroll
  for (int kt = 0; kt < 4; ++kt) {
    v16bf a = load_a_frag(rowp, kt * 32, lane);
#pragma unroll
    for (int nt = 0; nt < 8; ++nt) {
      v16bf b = load_b_frag(sW, kt, nt, lane);
      acc[nt] = wmma_bf16(a, b, acc[nt]);
    }
  }

  const int mh = (lane >> 4) * 8;
#pragma unroll
  for (int nt = 0; nt < 8; ++nt) {
    const int n  = nt * 16 + (lane & 15);
    const float bn = Ws_b[n];
#pragma unroll
    for (int v = 0; v < 8; ++v) {
      const int node = tb + mh + v;
      if (node < Nn)
        out[(size_t)node * D + n] =
            tanhf(acc[nt][v] + bn + agg[(size_t)node * D + n]);
    }
  }
}

// ---------------------------------------------------------------------------
// Host launch
// ---------------------------------------------------------------------------
extern "C" void kernel_launch(void* const* d_in, const int* in_sizes, int n_in,
                              void* d_out, int out_size, void* d_ws,
                              size_t ws_size, hipStream_t stream) {
  const float* node_emb = (const float*)d_in[0];
  const int*   er_src   = (const int*)d_in[1];
  const int*   er_dst   = (const int*)d_in[2];
  const int*   ee_src   = (const int*)d_in[3];
  // d_in[4] (ee_dst) unused: reference gathers AND scatters on ee_src
  const float* ee_w     = (const float*)d_in[5];
  const float* Wa_w     = (const float*)d_in[6];
  const float* Wa_b     = (const float*)d_in[7];
  const float* w0_w     = (const float*)d_in[8];
  const float* w0_b     = (const float*)d_in[9];
  const float* Ws_w     = (const float*)d_in[10];
  const float* Ws_b     = (const float*)d_in[11];
  const float* Wn_w     = (const float*)d_in[12];
  const float* Wn_b     = (const float*)d_in[13];
  (void)n_in; (void)out_size; (void)ws_size;

  const int Nn = in_sizes[0] / D;
  const int E1 = in_sizes[1];
  const int E2 = in_sizes[3];

  // workspace carve (~82 MB total)
  char* ws = (char*)d_ws;
  size_t off = 0;
  auto carve = [&](size_t bytes) -> char* {
    char* p = ws + off;
    off = (off + bytes + 255) & ~(size_t)255;
    return p;
  };
  unsigned short* node_bf = (unsigned short*)carve((size_t)Nn * D * 2);
  unsigned short* Wa_sw   = (unsigned short*)carve((size_t)2 * D * D * 2);
  unsigned short* Wn_sw   = (unsigned short*)carve((size_t)D * D * 2);
  unsigned short* Ws_sw   = (unsigned short*)carve((size_t)D * D * 2);
  float* evals  = (float*)carve((size_t)E1 * 4);
  float* segmax = (float*)carve((size_t)Nn * 4);
  float* segsum = (float*)carve((size_t)Nn * 4);
  float* agg    = (float*)carve((size_t)Nn * D * 4);

  // 1) bf16 conversion + B-fragment swizzle + zero init (deterministic)
  cvt_bf16_kernel<<<2048, 256, 0, stream>>>(node_emb, node_bf, Nn * D);
  swizzle_w_kernel<<<128, 256, 0, stream>>>(Wa_w, Wa_sw, 2 * D);
  swizzle_w_kernel<<<64, 256, 0, stream>>>(Wn_w, Wn_sw, D);
  swizzle_w_kernel<<<64, 256, 0, stream>>>(Ws_w, Ws_sw, D);
  zero_kernel<<<512, 256, 0, stream>>>(segmax, Nn);
  zero_kernel<<<512, 256, 0, stream>>>(segsum, Nn);
  zero_kernel<<<2048, 256, 0, stream>>>(agg, Nn * D);

  // 2) ER attention logits + segment max
  er_logits_kernel<<<(E1 + 127) / 128, 256, 0, stream>>>(
      node_bf, er_src, er_dst, E1, Wa_sw, Wa_b, w0_w, w0_b, evals, segmax);

  // 3) exp + segment sum
  expsum_kernel<<<(E1 + 255) / 256, 256, 0, stream>>>(segmax, evals, segsum,
                                                      er_src, E1);

  // 4) ER weighted message scatter (attn = exp / (sum + 1e-9))
  scatter_msg_kernel<<<(E1 + 127) / 128, 256, 0, stream>>>(
      node_bf, er_dst, er_src, E1, 0, evals, segsum, nullptr, Wn_sw, Wn_b, agg);

  // 5) EE weighted message scatter
  scatter_msg_kernel<<<(E2 + 127) / 128, 256, 0, stream>>>(
      node_bf, ee_src, ee_src, E2, 1, nullptr, nullptr, ee_w, Wn_sw, Wn_b, agg);

  // 6) self update + tanh
  final_kernel<<<(Nn + 127) / 128, 256, 0, stream>>>(node_bf, Ws_sw, Ws_b, agg,
                                                     (float*)d_out, Nn);
}
